// GCNLayer_5798205849837
// MI455X (gfx1250) — matrix operand order repaired and checked by reading, not compiled, not run on hardware
//
#include <hip/hip_runtime.h>
#include <hip/hip_bf16.h>

#define N_NODES 8192
#define NFEAT   128

typedef __bf16 bf16_t;
typedef __attribute__((ext_vector_type(16))) __bf16 v16bf;
typedef __attribute__((ext_vector_type(8)))  float  v8f;
typedef __attribute__((ext_vector_type(4)))  float  f32x4;
typedef int v4i __attribute__((vector_size(16)));  // pointee type the async builtin wants

union BF16x16 { v16bf v; uint4 q[2]; };

#define AS(n) __attribute__((address_space(n)))

// gfx1250 async global->LDS path (ASYNCcnt-tracked, bypasses VGPRs).
#if defined(__has_builtin)
#if __has_builtin(__builtin_amdgcn_global_load_async_to_lds_b128)
#define USE_ASYNC_LDS 1
#endif
#endif
#ifndef USE_ASYNC_LDS
#define USE_ASYNC_LDS 0
#endif

__device__ __forceinline__ void wait_asynccnt0() {
#if defined(__has_builtin) && __has_builtin(__builtin_amdgcn_s_wait_asynccnt)
  __builtin_amdgcn_s_wait_asynccnt(0);
#else
  asm volatile("s_wait_asynccnt 0x0" ::: "memory");
#endif
}

// Stage one 64-byte column chunk (4 x b128) global -> LDS.
__device__ __forceinline__ void stage_chunk(const bf16_t* g, unsigned char* l) {
#if USE_ASYNC_LDS
  AS(1) v4i* gp = (AS(1) v4i*)g;
  AS(3) v4i* lp = (AS(3) v4i*)l;
  __builtin_amdgcn_global_load_async_to_lds_b128(gp + 0, lp + 0, 0, 0);
  __builtin_amdgcn_global_load_async_to_lds_b128(gp + 1, lp + 1, 0, 0);
  __builtin_amdgcn_global_load_async_to_lds_b128(gp + 2, lp + 2, 0, 0);
  __builtin_amdgcn_global_load_async_to_lds_b128(gp + 3, lp + 3, 0, 0);
#else
  const uint4* src = (const uint4*)g;
  uint4* dst = (uint4*)l;
  dst[0] = src[0]; dst[1] = src[1]; dst[2] = src[2]; dst[3] = src[3];
#endif
}

// ---------------------------------------------------------------------------
// Kernel 1: dinv[r] = rsqrt(1 + sum_c A[r][c])   (A_hat = A + I row sums)
// Non-temporal b128 loads: 268 MB stream, never re-read by this kernel.
// ---------------------------------------------------------------------------
__global__ __launch_bounds__(256) void gcn_dinv_kernel(
    const float* __restrict__ A, float* __restrict__ dinv) {
  __shared__ float red[256];
  const int row = blockIdx.x;
  const f32x4* ap = (const f32x4*)(A + (size_t)row * N_NODES);
  float s = 0.0f;
#pragma unroll 4
  for (int c = threadIdx.x; c < N_NODES / 4; c += 256) {
    f32x4 v = __builtin_nontemporal_load(ap + c);
    s += (v.x + v.y) + (v.z + v.w);
  }
  red[threadIdx.x] = s;
  __syncthreads();
  for (int off = 128; off > 0; off >>= 1) {
    if (threadIdx.x < off) red[threadIdx.x] += red[threadIdx.x + off];
    __syncthreads();
  }
  if (threadIdx.x == 0) dinv[row] = rsqrtf(red[0] + 1.0f);
}

// ---------------------------------------------------------------------------
// Kernel 2: supportT[f][k] = dinv[k] * ( x[k,:] . W[f,:] + b[f] )  as bf16
// Stored TRANSPOSED so the main GEMM's B fragments are K-contiguous.
// ---------------------------------------------------------------------------
__global__ __launch_bounds__(NFEAT) void gcn_support_kernel(
    const float* __restrict__ x, const float* __restrict__ W,
    const float* __restrict__ b, const float* __restrict__ dinv,
    bf16_t* __restrict__ supportT) {
  __shared__ float xs[NFEAT];
  const int k = blockIdx.x;
  const int f = threadIdx.x;
  xs[f] = x[(size_t)k * NFEAT + f];
  __syncthreads();
  const float* wr = W + (size_t)f * NFEAT;
  float acc = b[f];
#pragma unroll 8
  for (int c = 0; c < NFEAT; ++c) acc += xs[c] * wr[c];
  supportT[(size_t)f * N_NODES + k] = (bf16_t)(acc * dinv[k]);
}

// ---------------------------------------------------------------------------
// Kernel 3: out = diag(dinv) * (A + I) @ (diag(dinv) * support)
//   M = 8192, N = 128, K = 8192. bf16 WMMA, f32 accumulate.
// Block = 2 waves (64 thr). Each wave owns a 32-aligned 32-row band =
// 2 M-tiles x 8 N-tiles -> 16 C tiles; every B fragment read from LDS feeds
// TWO WMMAs. B panel double-buffered via GLOBAL_LOAD_ASYNC_TO_LDS_B128,
// one s_wait_asynccnt + barrier per K-step. +I handled in a wave-uniform
// rare branch; dinv[row] applied once at the C store.
// ---------------------------------------------------------------------------
#define KTILE 32
#define WAVES 2
#define ROWS_PER_WAVE 32                     // two 16-row tiles, 32-aligned band
#define BLOCK_ROWS (WAVES * ROWS_PER_WAVE)   // 64
#define LDS_ROW_BYTES 80                     // 64B K-data + 16B pad (bank spread)

__global__ __launch_bounds__(WAVES * 32) void gcn_aggregate_kernel(
    const float* __restrict__ A, const bf16_t* __restrict__ supportT,
    const float* __restrict__ dinv, float* __restrict__ out) {
  __shared__ __align__(16) unsigned char bsm[2][NFEAT * LDS_ROW_BYTES];  // 20 KB

  const int tid  = threadIdx.x;   // 0..63
  const int wave = tid >> 5;
  const int lane = tid & 31;
  const int lrow = lane & 15;   // M (A/C) or N (B) index inside a tile
  const int half = lane >> 4;   // selects K sub-range / M upper half

  const int band    = blockIdx.x * BLOCK_ROWS + wave * ROWS_PER_WAVE;  // 32-aligned
  const int diag_kb = band;     // the ONE K-block holding BOTH tiles' diagonals
  const int m0 = band + lrow;            // strip 0 row for this lane
  const int m1 = band + 16 + lrow;       // strip 1 row for this lane
  const float* arow0 = A + (size_t)m0 * N_NODES;
  const float* arow1 = A + (size_t)m1 * N_NODES;

  // staging: thread t owns supportT columns t and t+64 (64 threads, 128 cols)
  const bf16_t* gsrc0 = supportT + (size_t)tid * N_NODES;
  const bf16_t* gsrc1 = supportT + (size_t)(tid + 64) * N_NODES;
  unsigned char* lcol0[2] = { &bsm[0][tid * LDS_ROW_BYTES],
                              &bsm[1][tid * LDS_ROW_BYTES] };
  unsigned char* lcol1[2] = { &bsm[0][(tid + 64) * LDS_ROW_BYTES],
                              &bsm[1][(tid + 64) * LDS_ROW_BYTES] };

  v8f acc[2][8] = {};

  // ---- prologue: stage panel 0 into buffer 0
  stage_chunk(gsrc0, lcol0[0]);
  stage_chunk(gsrc1, lcol1[0]);
#if USE_ASYNC_LDS
  wait_asynccnt0();
#endif
  __syncthreads();

  for (int kb = 0; kb < N_NODES; kb += KTILE) {
    const int cur = (kb >> 5) & 1;
    unsigned char* cbase = &bsm[cur][0];

    // ---- stage NEXT panel into the other buffer (no wait yet)
    if (kb + KTILE < N_NODES) {
      stage_chunk(gsrc0 + kb + KTILE, lcol0[cur ^ 1]);
      stage_chunk(gsrc1 + kb + KTILE, lcol1[cur ^ 1]);
      __builtin_prefetch(arow0 + kb + 2 * KTILE, 0, 3);
      __builtin_prefetch(arow1 + kb + 2 * KTILE, 0, 3);
    }

    // ---- A fragments (2 strips): 16x32 bf16 per ISA layout, NT f32 stream.
    // half==0 holds K {0..7, 16..23}; half==1 holds K {8..15, 24..31}.
    const int k0 = kb + 8 * half;        // -> v16bf[0..7]
    const int k1 = kb + 16 + 8 * half;   // -> v16bf[8..15]
    BF16x16 afrag[2];
    {
      union { f32x4 v[2]; float f[8]; } c0[2], c1[2];
      c0[0].v[0] = __builtin_nontemporal_load((const f32x4*)(arow0 + k0) + 0);
      c0[0].v[1] = __builtin_nontemporal_load((const f32x4*)(arow0 + k0) + 1);
      c1[0].v[0] = __builtin_nontemporal_load((const f32x4*)(arow0 + k1) + 0);
      c1[0].v[1] = __builtin_nontemporal_load((const f32x4*)(arow0 + k1) + 1);
      c0[1].v[0] = __builtin_nontemporal_load((const f32x4*)(arow1 + k0) + 0);
      c0[1].v[1] = __builtin_nontemporal_load((const f32x4*)(arow1 + k0) + 1);
      c1[1].v[0] = __builtin_nontemporal_load((const f32x4*)(arow1 + k1) + 0);
      c1[1].v[1] = __builtin_nontemporal_load((const f32x4*)(arow1 + k1) + 1);

      if (kb == diag_kb) {  // wave-uniform: taken once in 256 iterations
#pragma unroll
        for (int s = 0; s < 2; ++s) {
          const int m = band + 16 * s + lrow;
#pragma unroll
          for (int j = 0; j < 8; ++j) {
            afrag[s].v[j]     = (bf16_t)(c0[s].f[j] + ((k0 + j == m) ? 1.0f : 0.0f));
            afrag[s].v[8 + j] = (bf16_t)(c1[s].f[j] + ((k1 + j == m) ? 1.0f : 0.0f));
          }
        }
      } else {              // common path: pure packed converts
#pragma unroll
        for (int s = 0; s < 2; ++s) {
#pragma unroll
          for (int j = 0; j < 8; ++j) {
            afrag[s].v[j]     = (bf16_t)c0[s].f[j];
            afrag[s].v[8 + j] = (bf16_t)c1[s].f[j];
          }
        }
      }
    }

    // ---- 8 B fragments from LDS; each feeds TWO WMMAs (16 total).
    // B 32x16 bf16: lanes 0-15 K{0..7,16..23}, lanes 16-31 K{8..15,24..31}.
#pragma unroll
    for (int n = 0; n < 8; ++n) {
      BF16x16 bfrag;
      const uint4* bp =
          (const uint4*)(cbase + (n * 16 + lrow) * LDS_ROW_BYTES + half * 16);
      bfrag.q[0] = bp[0];
      bfrag.q[1] = bp[2];
      acc[0][n] = __builtin_amdgcn_wmma_f32_16x16x32_bf16(
          false, afrag[0].v, false, bfrag.v, (short)0, acc[0][n], false, false);
      acc[1][n] = __builtin_amdgcn_wmma_f32_16x16x32_bf16(
          false, afrag[1].v, false, bfrag.v, (short)0, acc[1][n], false, false);
    }

    // ---- single barrier per K-step: staging done + panel reads done
#if USE_ASYNC_LDS
    wait_asynccnt0();
#endif
    __syncthreads();
  }

  // ---- store C tiles, applying dinv[row] once here.
  // VGPR r -> row (row_base + r), col = n*16 + lrow.
#pragma unroll
  for (int s = 0; s < 2; ++s) {
    const int row_base = band + 16 * s + 8 * half;  // multiple of 8
    union { f32x4 v[2]; float f[8]; } dr;
    {
      const f32x4* dv = (const f32x4*)(dinv + row_base);
      dr.v[0] = dv[0];
      dr.v[1] = dv[1];
    }
#pragma unroll
    for (int n = 0; n < 8; ++n) {
#pragma unroll
      for (int r = 0; r < 8; ++r) {
        __builtin_nontemporal_store(
            acc[s][n][r] * dr.f[r],
            out + (size_t)(row_base + r) * NFEAT + n * 16 + lrow);
      }
    }
  }
}

// ---------------------------------------------------------------------------
extern "C" void kernel_launch(void* const* d_in, const int* in_sizes, int n_in,
                              void* d_out, int out_size, void* d_ws, size_t ws_size,
                              hipStream_t stream) {
  const float* x = (const float*)d_in[0];
  const float* A = (const float*)d_in[1];
  const float* W = (const float*)d_in[2];
  const float* b = (const float*)d_in[3];
  float* out = (float*)d_out;

  // workspace: [0, 32KB) dinv f32 ; [32KB, 32KB + 2MB) supportT bf16
  float* dinv = (float*)d_ws;
  bf16_t* supportT = (bf16_t*)((char*)d_ws + N_NODES * sizeof(float));

  gcn_dinv_kernel<<<N_NODES, 256, 0, stream>>>(A, dinv);
  gcn_support_kernel<<<N_NODES, NFEAT, 0, stream>>>(x, W, b, dinv, supportT);
  gcn_aggregate_kernel<<<N_NODES / BLOCK_ROWS, WAVES * 32, 0, stream>>>(
      A, supportT, dinv, out);
}
